// SJunNet12_46024869544415
// MI455X (gfx1250) — matrix operand order repaired and checked by reading, not compiled, not run on hardware
//
#include <hip/hip_runtime.h>
#include <hip/hip_bf16.h>

typedef _Float16 half_t;
typedef __attribute__((ext_vector_type(16))) _Float16 v16h;
typedef __attribute__((ext_vector_type(8)))  _Float16 v8h;
typedef __attribute__((ext_vector_type(8)))  float    v8f;

union VF  { v8f  v; float  f[8];  };
union H16 { v16h v; v8h    h[2];  };

__device__ __forceinline__ v8f wmma32(v16h a, v16h b, v8f c) {
  return __builtin_amdgcn_wmma_f32_16x16x32_f16(false, a, false, b, (short)0, c, false, false);
}

// A fragment: 16x32 f16, row-major A. lane (0-15): row = lane, K = kk+0..15
//             lane (16-31): same rows, K = kk+16..31
__device__ __forceinline__ v16h load_a_frag(const half_t* arow, int kk, int hi) {
  H16 u;
  const half_t* p = arow + kk + hi * 16;
  u.h[0] = *(const v8h*)(p);
  u.h[1] = *(const v8h*)(p + 8);
  return u.v;
}

// B fragment: 32x16 f16 from B^T stored row-major [N][K].
// lane (0-15): col = lane, K = kk+0..15 ; lane (16-31): col = lane-16, K = kk+16..31
__device__ __forceinline__ v16h load_b_frag(const half_t* bt, int ldb, int col0, int kk, int lane) {
  int n  = col0 + (lane & 15);
  int kb = kk + ((lane >> 4) << 4);
  const half_t* p = bt + (size_t)n * ldb + kb;
  H16 u;
  u.h[0] = *(const v8h*)(p);
  u.h[1] = *(const v8h*)(p + 8);
  return u.v;
}

// async copy of one 16-byte chunk: global -> LDS (ASYNCcnt tracked)
__device__ __forceinline__ void async_b128(unsigned lds_off, const void* gaddr) {
  unsigned long long ga = (unsigned long long)(uintptr_t)gaddr;
  asm volatile("global_load_async_to_lds_b128 %0, %1, off"
               :: "v"(lds_off), "v"(ga) : "memory");
}
__device__ __forceinline__ void wait_async0() {
  asm volatile("s_wait_asynccnt 0x0" ::: "memory");
}

// ---------------- weight convert: f32 [K][N] -> f16 transposed [N][K] ----------------
__global__ __launch_bounds__(256) void wconv_kernel(const float* __restrict__ in,
                                                    half_t* __restrict__ out, int K, int N) {
  int idx = (int)blockIdx.x * 256 + (int)threadIdx.x;
  if (idx >= K * N) return;
  int n = idx / K, k = idx % K;
  out[idx] = (half_t)in[(size_t)k * N + n];
}

// ---------------- LayerNorm1: x [C=256][65536] channel-major -> xn f16 [tok][256] ----
__global__ __launch_bounds__(256) void ln1_kernel(const float* __restrict__ x,
                                                  const float* __restrict__ gamma,
                                                  const float* __restrict__ beta,
                                                  half_t* __restrict__ xn) {
  __shared__ float tile[256 * 33];
  __shared__ float smean[32], srstd[32];
  int tid  = threadIdx.x;
  int lane = tid & 31, wv = tid >> 5;
  int g0   = (int)blockIdx.x * 32;
  for (int j = 0; j < 32; ++j) {
    int c = wv + 8 * j;
    tile[c * 33 + lane] = x[(size_t)c * 65536 + g0 + lane];
  }
  __syncthreads();
  {
    int tok = wv * 4 + (lane >> 3);
    int s   = lane & 7;
    float sum = 0.f, sq = 0.f;
    for (int c = s * 32; c < s * 32 + 32; ++c) {
      float v = tile[c * 33 + tok];
      sum += v; sq += v * v;
    }
    for (int m = 1; m < 8; m <<= 1) {
      sum += __shfl_xor(sum, m, 32);
      sq  += __shfl_xor(sq,  m, 32);
    }
    if (s == 0) {
      float mean = sum * (1.f / 256.f);
      float var  = sq * (1.f / 256.f) - mean * mean;
      smean[tok] = mean;
      srstd[tok] = rsqrtf(var + 1e-5f);
    }
  }
  __syncthreads();
  for (int j = 0; j < 32; ++j) {
    int idx = tid + 256 * j;
    int c = idx & 255, gg = idx >> 8;
    float v = (tile[c * 33 + gg] - smean[gg]) * srstd[gg] * gamma[c] + beta[c];
    xn[(size_t)(g0 + gg) * 256 + c] = (half_t)v;
  }
}

// ---------------- QKV GEMM: rows permuted to window order, q pre-scaled ----------------
__global__ __launch_bounds__(256) void qkv_kernel(const half_t* __restrict__ xn,
                                                  const half_t* __restrict__ wT,
                                                  half_t* __restrict__ out,
                                                  int chOff, int Wh, int Ww) {
  __shared__ __align__(16) half_t ctile[8 * 1024];        // 16x64 f16 per wave
  int waveId = (int)blockIdx.x * 8 + ((int)threadIdx.x >> 5);
  int wv     = (int)threadIdx.x >> 5;
  int lane   = (int)threadIdx.x & 31;
  int mTile  = waveId / 6, nTile = waveId % 6;
  int row0   = mTile * 16, col0 = nTile * 64;
  int hi     = lane >> 4;

  int r   = row0 + (lane & 15);
  int win = r >> 6, n = r & 63;
  int nWw = 256 / Ww;
  int wy = win / nWw, wx = win % nWw;
  int ii = n / Ww,   jj = n % Ww;
  int gpos = (wy * Wh + ii) * 256 + (wx * Ww + jj);
  const half_t* arow = xn + (size_t)gpos * 256 + chOff;

  VF acc[4];
  for (int t = 0; t < 4; ++t) for (int q = 0; q < 8; ++q) acc[t].f[q] = 0.f;
  for (int kk = 0; kk < 128; kk += 32) {
    v16h a = load_a_frag(arow, kk, hi);
    for (int t = 0; t < 4; ++t) {
      v16h b = load_b_frag(wT, 128, col0 + 16 * t, kk, lane);
      acc[t].v = wmma32(a, b, acc[t].v);
    }
  }
  int colbase = lane & 15;
  half_t* ct = ctile + wv * 1024;
  for (int t = 0; t < 4; ++t)
    for (int rr = 0; rr < 8; ++rr) {
      int c = 16 * t + colbase;
      float v = acc[t].f[rr];
      if (c + col0 < 128) v *= 0.125f;               // q * d^-0.5
      ct[(rr + 8 * hi) * 64 + c] = (half_t)v;
    }
  __syncthreads();
  // coalesced store: 32 lanes write one 128B row
  const unsigned* ct32 = (const unsigned*)ct;
  unsigned* o32 = (unsigned*)out;
  for (int j = 0; j < 16; ++j) {
    int m = row0 + j;
    o32[(size_t)m * 192 + (col0 >> 1) + lane] = ct32[j * 32 + lane];
  }
}

// ---------------- attention: one block per window (N=64 tokens, 2 heads, d=64) --------
#define VPAD 80
__global__ __launch_bounds__(256) void attn_kernel(const half_t* __restrict__ qkv,
                                                   const float* __restrict__ table,
                                                   half_t* __restrict__ attnout,
                                                   int chOffOut, int Wh, int Ww) {
  __shared__ __align__(16) half_t vT[2 * 64 * VPAD];
  __shared__ __align__(16) half_t pP[2 * 64 * VPAD];
  __shared__ __align__(16) half_t ctile[8 * 1024];
  int win  = (int)blockIdx.x;
  size_t base = (size_t)win * 64;
  int tid  = threadIdx.x;
  int lane = tid & 31, wv = tid >> 5;

  // stage V^T per head into LDS: vT[h][e][tok]
  for (int j = 0; j < 32; ++j) {
    int linear = tid + 256 * j;            // 8192 = 64 tok * 128 e
    int ef  = linear & 127;
    int tok = linear >> 7;
    int h   = ef >> 6, e = ef & 63;
    vT[(h * 64 + e) * VPAD + tok] = qkv[(base + tok) * 384 + 256 + ef];
  }
  __syncthreads();

  int h     = wv >> 2;
  int strip = wv & 3;
  int row0  = strip * 16;
  int hi    = lane >> 4;
  int colbase = lane & 15;

  // ---- S = q k^T (q pre-scaled) ----
  const half_t* arow  = qkv + (base + row0 + (lane & 15)) * 384 + h * 64;
  const half_t* bbase = qkv + base * 384 + 128 + h * 64;     // k rows, ldb = 384
  VF acc[4];
  for (int t = 0; t < 4; ++t) for (int q = 0; q < 8; ++q) acc[t].f[q] = 0.f;
  for (int kk = 0; kk < 64; kk += 32) {
    v16h a = load_a_frag(arow, kk, hi);
    for (int t = 0; t < 4; ++t) {
      v16h b = load_b_frag(bbase, 384, 16 * t, kk, lane);
      acc[t].v = wmma32(a, b, acc[t].v);
    }
  }
  // ---- + relative position bias ----
  for (int t = 0; t < 4; ++t)
    for (int rr = 0; rr < 8; ++rr) {
      int q = row0 + rr + 8 * hi;
      int k = 16 * t + colbase;
      int qi = q / Ww, qj = q % Ww, ki = k / Ww, kj = k % Ww;
      int idx = (qi - ki + Wh - 1) * (2 * Ww - 1) + (qj - kj + Ww - 1);
      acc[t].f[rr] += table[idx * 2 + h];
    }
  // ---- softmax over 64 keys ----
  for (int rr = 0; rr < 8; ++rr) {
    float m = fmaxf(fmaxf(acc[0].f[rr], acc[1].f[rr]), fmaxf(acc[2].f[rr], acc[3].f[rr]));
    for (int msk = 1; msk < 16; msk <<= 1) m = fmaxf(m, __shfl_xor(m, msk, 32));
    float s = 0.f;
    for (int t = 0; t < 4; ++t) {
      acc[t].f[rr] = expf(acc[t].f[rr] - m);
      s += acc[t].f[rr];
    }
    for (int msk = 1; msk < 16; msk <<= 1) s += __shfl_xor(s, msk, 32);
    float inv = 1.f / s;
    int q = row0 + rr + 8 * hi;
    for (int t = 0; t < 4; ++t)
      pP[(h * 64 + q) * VPAD + 16 * t + colbase] = (half_t)(acc[t].f[rr] * inv);
  }
  __syncthreads();
  // ---- O = P V (both operands from LDS) ----
  const half_t* parow = pP + (h * 64 + row0 + (lane & 15)) * VPAD;
  const half_t* vbase = vT + (h * 64) * VPAD;
  VF o[4];
  for (int t = 0; t < 4; ++t) for (int q = 0; q < 8; ++q) o[t].f[q] = 0.f;
  for (int kk = 0; kk < 64; kk += 32) {
    v16h a = load_a_frag(parow, kk, hi);
    for (int t = 0; t < 4; ++t) {
      v16h b = load_b_frag(vbase, VPAD, 16 * t, kk, lane);
      o[t].v = wmma32(a, b, o[t].v);
    }
  }
  half_t* ct = ctile + wv * 1024;
  for (int t = 0; t < 4; ++t)
    for (int rr = 0; rr < 8; ++rr)
      ct[(rr + 8 * hi) * 64 + 16 * t + colbase] = (half_t)o[t].f[rr];
  __syncthreads();
  const unsigned* ct32 = (const unsigned*)ct;
  unsigned* o32 = (unsigned*)attnout;
  int cword = (chOffOut + h * 64) >> 1;
  for (int j = 0; j < 16; ++j) {
    size_t m = base + row0 + j;
    o32[m * 128 + cword + lane] = ct32[j * 32 + lane];
  }
}

// ---------------- proj GEMM + (idt + xn + proj+bias), un-permute to natural order -----
__global__ __launch_bounds__(256) void proj_kernel(const half_t* __restrict__ attn,
                                                   const half_t* __restrict__ wT,
                                                   const float* __restrict__ bias,
                                                   const half_t* __restrict__ xn,
                                                   const float* __restrict__ xin,
                                                   float* __restrict__ x2,
                                                   int chOff, int Wh, int Ww) {
  int waveId = (int)blockIdx.x * 8 + ((int)threadIdx.x >> 5);
  int lane   = (int)threadIdx.x & 31;
  int mTile  = waveId >> 1, nTile = waveId & 1;
  int row0   = mTile * 16, col0 = nTile * 64;
  int hi     = lane >> 4;

  const half_t* arow = attn + (size_t)(row0 + (lane & 15)) * 256 + chOff;
  VF acc[4];
  for (int t = 0; t < 4; ++t) for (int q = 0; q < 8; ++q) acc[t].f[q] = 0.f;
  for (int kk = 0; kk < 128; kk += 32) {
    v16h a = load_a_frag(arow, kk, hi);
    for (int t = 0; t < 4; ++t) {
      v16h b = load_b_frag(wT, 128, col0 + 16 * t, kk, lane);
      acc[t].v = wmma32(a, b, acc[t].v);
    }
  }
  int nWw = 256 / Ww;
  int colbase = lane & 15;
  for (int t = 0; t < 4; ++t)
    for (int rr = 0; rr < 8; ++rr) {
      int m = row0 + rr + 8 * hi;
      int c = col0 + 16 * t + colbase;
      int win = m >> 6, n = m & 63;
      int wy = win / nWw, wx = win % nWw;
      int ii = n / Ww, jj = n % Ww;
      int g  = (wy * Wh + ii) * 256 + (wx * Ww + jj);
      float v = acc[t].f[rr] + bias[c]
              + (float)xn[(size_t)g * 256 + chOff + c]
              + xin[(size_t)(chOff + c) * 65536 + g];
      x2[(size_t)g * 256 + chOff + c] = v;
    }
}

// ---------------- LayerNorm2: x2 [tok][256] f32 -> f16 ----------------
__global__ __launch_bounds__(256) void ln2_kernel(const float* __restrict__ x2,
                                                  const float* __restrict__ gamma,
                                                  const float* __restrict__ beta,
                                                  half_t* __restrict__ out) {
  int row  = (int)blockIdx.x * 8 + ((int)threadIdx.x >> 5);
  int lane = (int)threadIdx.x & 31;
  const float* xr = x2 + (size_t)row * 256;
  float sum = 0.f, sq = 0.f;
  for (int j = 0; j < 8; ++j) {
    float v = xr[lane + 32 * j];
    sum += v; sq += v * v;
  }
  for (int m = 1; m < 32; m <<= 1) {
    sum += __shfl_xor(sum, m, 32);
    sq  += __shfl_xor(sq,  m, 32);
  }
  float mean = sum * (1.f / 256.f);
  float rstd = rsqrtf(sq * (1.f / 256.f) - mean * mean + 1e-5f);
  for (int j = 0; j < 8; ++j) {
    int c = lane + 32 * j;
    out[(size_t)row * 256 + c] = (half_t)((xr[c] - mean) * rstd * gamma[c] + beta[c]);
  }
}

// ---------------- MLP1: [65536,256] x [256,1024] + bias + exact GELU ----------------
// all 8 waves of a block share one 16-row A tile -> async-stage it in LDS once
__global__ __launch_bounds__(256) void mlp1_kernel(const half_t* __restrict__ a,
                                                   const half_t* __restrict__ wT,
                                                   const float* __restrict__ bias,
                                                   half_t* __restrict__ hid) {
  __shared__ __align__(16) half_t Atile[16 * 256];        // 8KB
  __shared__ __align__(16) half_t ctile[8 * 1024];        // 16KB
  int tid  = threadIdx.x;
  int lane = tid & 31, wv = tid >> 5;
  int mTile = (int)blockIdx.x >> 1;
  int nTile = (((int)blockIdx.x & 1) << 3) + wv;
  int row0  = mTile * 16, col0 = nTile * 64;
  int hi    = lane >> 4;

  unsigned abase = (unsigned)(uintptr_t)&Atile[0];
  const char* gsrc = (const char*)(a + (size_t)row0 * 256);
  for (int j = 0; j < 2; ++j) {                 // 16 rows * 512B = 512 x 16B chunks
    int chunk = tid + 256 * j;
    async_b128(abase + chunk * 16, gsrc + (size_t)chunk * 16);
  }
  wait_async0();
  __syncthreads();

  const half_t* arow = Atile + (lane & 15) * 256;
  VF acc[4];
  for (int t = 0; t < 4; ++t) for (int q = 0; q < 8; ++q) acc[t].f[q] = 0.f;
  for (int kk = 0; kk < 256; kk += 32) {
    v16h af = load_a_frag(arow, kk, hi);
    for (int t = 0; t < 4; ++t) {
      v16h b = load_b_frag(wT, 256, col0 + 16 * t, kk, lane);
      acc[t].v = wmma32(af, b, acc[t].v);
    }
  }
  int colbase = lane & 15;
  half_t* ct = ctile + wv * 1024;
  for (int t = 0; t < 4; ++t)
    for (int rr = 0; rr < 8; ++rr) {
      int c = col0 + 16 * t + colbase;
      float v = acc[t].f[rr] + bias[c];
      v = 0.5f * v * (1.f + erff(v * 0.70710678118654752f));   // exact GELU
      ct[(rr + 8 * hi) * 64 + 16 * t + colbase] = (half_t)v;
    }
  __syncthreads();
  const unsigned* ct32 = (const unsigned*)ct;
  unsigned* o32 = (unsigned*)hid;
  for (int j = 0; j < 16; ++j) {
    int m = row0 + j;
    o32[(size_t)m * 512 + (col0 >> 1) + lane] = ct32[j * 32 + lane];
  }
}

// ---------------- MLP2: [65536,1024] x [1024,256] + bias + residual -> [C][H*W] ------
// 128-thread blocks: 4 waves = 4 N-tiles sharing one async-staged 16x1024 A tile
__global__ __launch_bounds__(128) void mlp2_kernel(const half_t* __restrict__ hid,
                                                   const half_t* __restrict__ wT,
                                                   const float* __restrict__ bias,
                                                   const float* __restrict__ x2,
                                                   float* __restrict__ dout) {
  __shared__ __align__(16) half_t Atile[16 * 1024];       // 32KB
  int tid  = threadIdx.x;
  int lane = tid & 31, wv = tid >> 5;
  int row0 = (int)blockIdx.x * 16;
  int col0 = wv * 64;
  int hi   = lane >> 4;

  unsigned abase = (unsigned)(uintptr_t)&Atile[0];
  const char* gsrc = (const char*)(hid + (size_t)row0 * 1024);
  for (int j = 0; j < 16; ++j) {                // 16 rows * 2KB = 2048 x 16B chunks
    int chunk = tid + 128 * j;
    async_b128(abase + chunk * 16, gsrc + (size_t)chunk * 16);
  }
  wait_async0();
  __syncthreads();

  const half_t* arow = Atile + (lane & 15) * 1024;
  VF acc[4];
  for (int t = 0; t < 4; ++t) for (int q = 0; q < 8; ++q) acc[t].f[q] = 0.f;
  for (int kk = 0; kk < 1024; kk += 32) {
    v16h af = load_a_frag(arow, kk, hi);
    for (int t = 0; t < 4; ++t) {
      v16h b = load_b_frag(wT, 1024, col0 + 16 * t, kk, lane);
      acc[t].v = wmma32(af, b, acc[t].v);
    }
  }
  int colbase = lane & 15;
  for (int t = 0; t < 4; ++t)
    for (int rr = 0; rr < 8; ++rr) {
      int m = row0 + rr + 8 * hi;
      int c = col0 + 16 * t + colbase;
      float v = acc[t].f[rr] + bias[c] + x2[(size_t)m * 256 + c];
      dout[(size_t)c * 65536 + m] = v;          // transpose back to [B,C,H,W]
    }
}

// =====================================================================================
extern "C" void kernel_launch(void* const* d_in, const int* in_sizes, int n_in,
                              void* d_out, int out_size, void* d_ws, size_t ws_size,
                              hipStream_t stream) {
  (void)in_sizes; (void)n_in; (void)out_size; (void)ws_size;
  const float* x        = (const float*)d_in[0];
  const float* n1g      = (const float*)d_in[1];
  const float* n1b      = (const float*)d_in[2];
  const float* qkv_r    = (const float*)d_in[3];
  const float* proj_r_w = (const float*)d_in[4];
  const float* proj_r_b = (const float*)d_in[5];
  const float* table_r  = (const float*)d_in[6];
  const float* qkv_a    = (const float*)d_in[7];
  const float* proj_a_w = (const float*)d_in[8];
  const float* proj_a_b = (const float*)d_in[9];
  const float* table_a  = (const float*)d_in[10];
  const float* n2g      = (const float*)d_in[11];
  const float* n2b      = (const float*)d_in[12];
  const float* w1       = (const float*)d_in[13];
  const float* b1       = (const float*)d_in[14];
  const float* w2       = (const float*)d_in[15];
  const float* b2       = (const float*)d_in[16];
  float* out            = (float*)d_out;

  const size_t MB = 1024ull * 1024ull;
  char* base = (char*)d_ws;
  half_t* XN   = (half_t*)(base + 0);          // [65536][256] f16, 32MB
  half_t* QKVR = (half_t*)(base + 32  * MB);   // [65536][384] f16, 48MB
  half_t* QKVA = (half_t*)(base + 80  * MB);   // [65536][384] f16, 48MB
  half_t* ATTN = (half_t*)(base + 128 * MB);   // [65536][256] f16, 32MB
  float*  X2   = (float*) (base + 32  * MB);   // reuses QKVR/QKVA region (free by then)
  half_t* LN2  = (half_t*)(base + 96  * MB);   // reuses tail of QKVA region
  half_t* HID  = (half_t*)(base + 128 * MB);   // [65536][1024] f16, reuses ATTN region
  half_t* WTS  = (half_t*)(base + 256 * MB);
  half_t* wqkvrT  = WTS;                       // [384][128]
  half_t* wprojrT = WTS + 49152;               // [128][128]
  half_t* wqkvaT  = WTS + 65536;               // [384][128]
  half_t* wprojaT = WTS + 114688;              // [128][128]
  half_t* w1T     = WTS + 131072;              // [1024][256]
  half_t* w2T     = WTS + 393216;              // [256][1024]

  // weight conversion (f32 row-major [K][N] -> f16 transposed [N][K])
  wconv_kernel<<<192,  256, 0, stream>>>(qkv_r,    wqkvrT, 128, 384);
  wconv_kernel<<<64,   256, 0, stream>>>(proj_r_w, wprojrT, 128, 128);
  wconv_kernel<<<192,  256, 0, stream>>>(qkv_a,    wqkvaT, 128, 384);
  wconv_kernel<<<64,   256, 0, stream>>>(proj_a_w, wprojaT, 128, 128);
  wconv_kernel<<<1024, 256, 0, stream>>>(w1,       w1T, 256, 1024);
  wconv_kernel<<<1024, 256, 0, stream>>>(w2,       w2T, 1024, 256);

  ln1_kernel<<<2048, 256, 0, stream>>>(x, n1g, n1b, XN);

  qkv_kernel<<<3072, 256, 0, stream>>>(XN, wqkvrT, QKVR, 0,   16, 4);
  qkv_kernel<<<3072, 256, 0, stream>>>(XN, wqkvaT, QKVA, 128, 4, 16);

  attn_kernel<<<1024, 256, 0, stream>>>(QKVR, table_r, ATTN, 0,   16, 4);
  attn_kernel<<<1024, 256, 0, stream>>>(QKVA, table_a, ATTN, 128, 4, 16);

  proj_kernel<<<1024, 256, 0, stream>>>(ATTN, wprojrT, proj_r_b, XN, x, X2, 0,   16, 4);
  proj_kernel<<<1024, 256, 0, stream>>>(ATTN, wprojaT, proj_a_b, XN, x, X2, 128, 4, 16);

  ln2_kernel<<<8192, 256, 0, stream>>>(X2, n2g, n2b, LN2);

  mlp1_kernel<<<8192, 256, 0, stream>>>(LN2, w1T, b1, HID);
  mlp2_kernel<<<4096, 128, 0, stream>>>(HID, w2T, b2, X2, out);
}
